// VectorQuantizer_49538152792117
// MI455X (gfx1250) — matrix operand order repaired and checked by reading, not compiled, not run on hardware
//
#include <hip/hip_runtime.h>
#include <hip/hip_bf16.h>

typedef __attribute__((ext_vector_type(16))) __bf16 v16bf;
typedef __attribute__((ext_vector_type(8)))  float  v8f;

#define N_TOK     32768      // 8*64*64 tokens
#define C_DIM     256
#define K_CODES   8192
#define M_TILES   2048       // N_TOK/16
#define N_TILES   512        // K_CODES/16
#define KSTEPS    8          // C_DIM/32
#define HW        4096       // 64*64
#define OUT_ELEMS 33554432   // 8*256*64*64
#define GL_BLOCKS 16384      // gather/loss blocks (256 thr * 8 elem)

// workspace layout (bytes)
#define WS_ZB   ((size_t)0)                      // 32768*256*2  = 16 MiB  (bf16 A-frags)
#define WS_EB   ((size_t)16u * 1024u * 1024u)    //  8192*256*2  =  4 MiB  (bf16 B-frags)
#define WS_ESQ  (WS_EB  + (size_t)4u * 1024u * 1024u)   // 8192*4
#define WS_IDX  (WS_ESQ + (size_t)32u * 1024u)          // 32768*4
#define WS_PART (WS_IDX + (size_t)128u * 1024u)         // 16384*4

__device__ __forceinline__ unsigned short f2bf(float f) {
    unsigned u = __float_as_uint(f);
    u += 0x7FFFu + ((u >> 16) & 1u);       // round-to-nearest-even
    return (unsigned short)(u >> 16);
}

// ---------------------------------------------------------------------------
// Pack z (B,C,H,W) -> bf16 A-fragments. 1 block per 16-token tile.
// Thread = (ks = tid>>5, lane = tid&31); writes its 16 bf16 (32B) lane slice.
// A 16x32 bf16 layout: lanes 0-15 (M=lane): K = ks*32 + {0..7,16..23};
//                      lanes 16-31        : K = ks*32 + {8..15,24..31}.
// ---------------------------------------------------------------------------
__global__ void vq_pack_z(const float* __restrict__ z,
                          unsigned short* __restrict__ zb) {
    int tile = blockIdx.x;
    int ks   = threadIdx.x >> 5;
    int lane = threadIdx.x & 31;
    int half = lane >> 4;
    int m    = lane & 15;
    int n    = tile * 16 + m;
    int b    = n >> 12;                 // /4096
    int hw   = n & (HW - 1);
    const float* zp = z + (size_t)b * C_DIM * HW + hw;
    int cbase = ks * 32 + half * 8;

    alignas(16) unsigned short tmp[16];
#pragma unroll
    for (int i = 0; i < 8; ++i)
        tmp[i] = f2bf(zp[(size_t)(cbase + i) * HW]);
#pragma unroll
    for (int i = 0; i < 8; ++i)
        tmp[8 + i] = f2bf(zp[(size_t)(cbase + 16 + i) * HW]);

    unsigned short* dst = zb + ((size_t)(tile * KSTEPS + ks) * 32 + lane) * 16;
    *(uint4*)(dst)     = *(const uint4*)(tmp);
    *(uint4*)(dst + 8) = *(const uint4*)(tmp + 8);
}

// ---------------------------------------------------------------------------
// Pack emb (K,C) -> bf16 B-fragments. 1 block per 16-code tile.
// B 32x16 bf16 layout: lanes 0-15 (N=lane): K = ks*32 + {0..7,16..23};
//                      lanes 16-31        : K = ks*32 + {8..15,24..31}.
// ---------------------------------------------------------------------------
__global__ void vq_pack_e(const float* __restrict__ emb,
                          unsigned short* __restrict__ eb) {
    int tile = blockIdx.x;
    int ks   = threadIdx.x >> 5;
    int lane = threadIdx.x & 31;
    int half = lane >> 4;
    int col  = lane & 15;
    int code = tile * 16 + col;
    const float* ep = emb + (size_t)code * C_DIM + ks * 32 + half * 8;

    alignas(16) unsigned short tmp[16];
#pragma unroll
    for (int i = 0; i < 8; ++i) tmp[i] = f2bf(ep[i]);
#pragma unroll
    for (int i = 0; i < 8; ++i) tmp[8 + i] = f2bf(ep[16 + i]);

    unsigned short* dst = eb + ((size_t)(tile * KSTEPS + ks) * 32 + lane) * 16;
    *(uint4*)(dst)     = *(const uint4*)(tmp);
    *(uint4*)(dst + 8) = *(const uint4*)(tmp + 8);
}

// ---------------------------------------------------------------------------
// e_sq[k] = sum_c emb[k][c]^2   (fp32; z^2 term dropped — constant per row)
// ---------------------------------------------------------------------------
__global__ void vq_esq(const float* __restrict__ emb, float* __restrict__ esq) {
    int code = blockIdx.x * 256 + threadIdx.x;
    const float4* ep = (const float4*)(emb + (size_t)code * C_DIM);
    float s = 0.f;
#pragma unroll 8
    for (int i = 0; i < C_DIM / 4; ++i) {
        float4 v = ep[i];
        s += v.x * v.x + v.y * v.y + v.z * v.z + v.w * v.w;
    }
    esq[code] = s;
}

// ---------------------------------------------------------------------------
// GEMM + argmin. 4 waves/block; each wave owns TWO 16-token tiles and keeps
// both 16x256 A operands resident in VGPRs (128 VGPRs), so every B fragment
// pulled from LDS feeds two v_wmma_f32_16x16x32_bf16. B tiles (16 codes x
// 256 dims, 8 KB) stream through double-buffered LDS with early global loads
// + global_prefetch_b8. Per-lane running min, wave32 shfl_xor reduce per half.
// ---------------------------------------------------------------------------
__global__ __launch_bounds__(128)
void vq_gemm_argmin(const unsigned short* __restrict__ zb,
                    const unsigned short* __restrict__ eb,
                    const float* __restrict__ esq,
                    int* __restrict__ idxOut) {
    __shared__ alignas(32) unsigned short lds[2][KSTEPS * 32 * 16]; // 2 x 8 KB

    int wave  = threadIdx.x >> 5;
    int lane  = threadIdx.x & 31;
    int tile0 = (blockIdx.x * 4 + wave) * 2;
    int tile1 = tile0 + 1;

    // Resident A fragments for this wave's 32 tokens (2 x 64 VGPRs)
    v16bf A0[KSTEPS], A1[KSTEPS];
    const v16bf* zf = (const v16bf*)zb;
#pragma unroll
    for (int ks = 0; ks < KSTEPS; ++ks) {
        A0[ks] = zf[(size_t)(tile0 * KSTEPS + ks) * 32 + lane];
        A1[ks] = zf[(size_t)(tile1 * KSTEPS + ks) * 32 + lane];
    }

    const uint4* ebv = (const uint4*)eb;          // 16-byte units
    const unsigned ent0 = threadIdx.x;            // each thread stages two 32B slices
    const unsigned ent1 = threadIdx.x + 128;

    // Preload code-tile 0 into LDS buffer 0
    {
        size_t b0 = ((size_t)0 * 256 + ent0) * 2;
        size_t b1 = ((size_t)0 * 256 + ent1) * 2;
        uint4 g0 = ebv[b0], g1 = ebv[b0 + 1];
        uint4 g2 = ebv[b1], g3 = ebv[b1 + 1];
        uint4* l0 = (uint4*)&lds[0][ent0 * 16];
        uint4* l1 = (uint4*)&lds[0][ent1 * 16];
        l0[0] = g0; l0[1] = g1;
        l1[0] = g2; l1[1] = g3;
    }
    __syncthreads();

    float best0[8], best1[8];
    int   bid0[8],  bid1[8];
#pragma unroll
    for (int j = 0; j < 8; ++j) {
        best0[j] = 3.4e38f; bid0[j] = 0;
        best1[j] = 3.4e38f; bid1[j] = 0;
    }

    for (int ct = 0; ct < N_TILES; ++ct) {
        const int  cur  = ct & 1;
        const bool have = (ct + 1) < N_TILES;

        // Issue next tile's global loads early (hidden under WMMA)
        uint4 g0, g1, g2, g3;
        if (have) {
            size_t b0 = ((size_t)(ct + 1) * 256 + ent0) * 2;
            size_t b1 = ((size_t)(ct + 1) * 256 + ent1) * 2;
            g0 = ebv[b0]; g1 = ebv[b0 + 1];
            g2 = ebv[b1]; g3 = ebv[b1 + 1];
        }
        if (ct + 2 < N_TILES)
            __builtin_prefetch(&ebv[((size_t)(ct + 2) * 256 + ent0) * 2], 0, 0);

        // Two 16x16 output tiles; each B fragment feeds two WMMAs (K = 256)
        v8f acc0 = {}, acc1 = {};
        const v16bf* lb = (const v16bf*)&lds[cur][0];
#pragma unroll
        for (int ks = 0; ks < KSTEPS; ++ks) {
            v16bf B = lb[ks * 32 + lane];
            acc0 = __builtin_amdgcn_wmma_f32_16x16x32_bf16(
                false, A0[ks], false, B, (short)0, acc0, false, false);
            acc1 = __builtin_amdgcn_wmma_f32_16x16x32_bf16(
                false, A1[ks], false, B, (short)0, acc1, false, false);
        }

        // score = e^2 - 2*(z.e); each lane owns column N = lane&15
        int   code = ct * 16 + (lane & 15);
        float es   = esq[code];
#pragma unroll
        for (int j = 0; j < 8; ++j) {
            float s0 = __builtin_fmaf(-2.0f, acc0[j], es);
            float s1 = __builtin_fmaf(-2.0f, acc1[j], es);
            if (s0 < best0[j]) { best0[j] = s0; bid0[j] = code; }
            if (s1 < best1[j]) { best1[j] = s1; bid1[j] = code; }
        }

        // Commit next tile into the other LDS buffer
        if (have) {
            uint4* l0 = (uint4*)&lds[cur ^ 1][ent0 * 16];
            uint4* l1 = (uint4*)&lds[cur ^ 1][ent1 * 16];
            l0[0] = g0; l0[1] = g1;
            l1[0] = g2; l1[1] = g3;
        }
        __syncthreads();
    }

    // Min-reduce across the 16 lanes of each half (xor<=8 stays in half);
    // tie-break to lower code index to match first-argmin semantics.
#pragma unroll
    for (int off = 8; off >= 1; off >>= 1) {
#pragma unroll
        for (int j = 0; j < 8; ++j) {
            float os0 = __shfl_xor(best0[j], off, 32);
            int   oi0 = __shfl_xor(bid0[j],  off, 32);
            if (os0 < best0[j] || (os0 == best0[j] && oi0 < bid0[j])) {
                best0[j] = os0; bid0[j] = oi0;
            }
            float os1 = __shfl_xor(best1[j], off, 32);
            int   oi1 = __shfl_xor(bid1[j],  off, 32);
            if (os1 < best1[j] || (os1 == best1[j] && oi1 < bid1[j])) {
                best1[j] = os1; bid1[j] = oi1;
            }
        }
    }
    if ((lane & 15) == 0) {
        int rb0 = tile0 * 16 + (lane >> 4) * 8;  // lanes<16: M=0..7, lanes>=16: M=8..15
        int rb1 = tile1 * 16 + (lane >> 4) * 8;
#pragma unroll
        for (int j = 0; j < 8; ++j) {
            idxOut[rb0 + j] = bid0[j];
            idxOut[rb1 + j] = bid1[j];
        }
    }
}

// ---------------------------------------------------------------------------
// Gather emb[idx] into NCHW output (straight-through value z + (q - z)),
// and per-block partial sums of (z - q)^2 for the loss.
// ---------------------------------------------------------------------------
__global__ void vq_gather_loss(const float* __restrict__ z,
                               const float* __restrict__ emb,
                               const int* __restrict__ idx,
                               float* __restrict__ out,
                               float* __restrict__ partial) {
    __shared__ float red[256];
    size_t base = (size_t)blockIdx.x * 2048 + threadIdx.x;
    float local = 0.f;
#pragma unroll
    for (int i = 0; i < 8; ++i) {
        size_t e  = base + (size_t)i * 256;       // output element (NCHW flat)
        int    c  = (int)((e >> 12) & 255);
        size_t n  = ((e >> 20) << 12) | (e & 4095);   // token = b*4096 + hw
        float  q  = emb[(size_t)idx[n] * C_DIM + c];
        float  zv = z[e];
        out[e] = zv + (q - zv);                   // match z + sg(q - z) rounding
        float d = zv - q;
        local += d * d;
    }
    red[threadIdx.x] = local;
    __syncthreads();
    for (int s = 128; s > 0; s >>= 1) {
        if (threadIdx.x < (unsigned)s) red[threadIdx.x] += red[threadIdx.x + s];
        __syncthreads();
    }
    if (threadIdx.x == 0) partial[blockIdx.x] = red[0];
}

// Deterministic final reduction (fixed order, no float atomics).
__global__ void vq_reduce(const float* __restrict__ partial,
                          float* __restrict__ out) {
    __shared__ float red[256];
    float s = 0.f;
    for (int i = threadIdx.x; i < GL_BLOCKS; i += 256) s += partial[i];
    red[threadIdx.x] = s;
    __syncthreads();
    for (int st = 128; st > 0; st >>= 1) {
        if (threadIdx.x < (unsigned)st) red[threadIdx.x] += red[threadIdx.x + st];
        __syncthreads();
    }
    if (threadIdx.x == 0)
        out[OUT_ELEMS] = 1.25f * red[0] / (float)OUT_ELEMS;  // (1 + 0.25) * mean
}

extern "C" void kernel_launch(void* const* d_in, const int* in_sizes, int n_in,
                              void* d_out, int out_size, void* d_ws, size_t ws_size,
                              hipStream_t stream) {
    (void)in_sizes; (void)n_in; (void)out_size; (void)ws_size;
    const float* z   = (const float*)d_in[0];   // (8,256,64,64) f32
    const float* emb = (const float*)d_in[1];   // (8192,256)    f32
    float* out = (float*)d_out;                 // 33554432 f32 + 1 loss

    char* ws = (char*)d_ws;
    unsigned short* zb  = (unsigned short*)(ws + WS_ZB);
    unsigned short* eb  = (unsigned short*)(ws + WS_EB);
    float*          esq = (float*)(ws + WS_ESQ);
    int*            idx = (int*)(ws + WS_IDX);
    float*          prt = (float*)(ws + WS_PART);

    vq_pack_z     <<<M_TILES,       256, 0, stream>>>(z, zb);
    vq_pack_e     <<<N_TILES,       256, 0, stream>>>(emb, eb);
    vq_esq        <<<K_CODES / 256, 256, 0, stream>>>(emb, esq);
    vq_gemm_argmin<<<M_TILES / 8,   128, 0, stream>>>(zb, eb, esq, idx);
    vq_gather_loss<<<GL_BLOCKS,     256, 0, stream>>>(z, emb, idx, out, prt);
    vq_reduce     <<<1,             256, 0, stream>>>(prt, out);
}
